// Encoder_27290222199159
// MI455X (gfx1250) — matrix operand order repaired and checked by reading, not compiled, not run on hardware
//
#include <hip/hip_runtime.h>
#include <hip/hip_bf16.h>

typedef __attribute__((ext_vector_type(16))) _Float16 v16h;
typedef __attribute__((ext_vector_type(8)))  float    v8f;
typedef __attribute__((ext_vector_type(4)))  unsigned int u32x4;
typedef __attribute__((ext_vector_type(8)))  int      i32x8;
typedef __attribute__((ext_vector_type(4)))  int      i32x4;

// ---------------------------------------------------------------------------
// conv0: 3->32, k4 s2 p1, direct fp32, relu -> f16 NCHW
// ---------------------------------------------------------------------------
__global__ void conv0_kernel(const float* __restrict__ x,
                             const float* __restrict__ w,
                             const float* __restrict__ bias,
                             _Float16* __restrict__ out)
{
    int idx = blockIdx.x * blockDim.x + threadIdx.x;
    if (idx >= 64 * 32 * 128 * 128) return;
    int ow = idx & 127;
    int oh = (idx >> 7) & 127;
    int oc = (idx >> 14) & 31;
    int b  = idx >> 19;
    float acc = bias[oc];
#pragma unroll
    for (int ic = 0; ic < 3; ++ic) {
#pragma unroll
        for (int kh = 0; kh < 4; ++kh) {
            int ih = oh * 2 - 1 + kh;
            if (ih < 0 || ih >= 256) continue;
#pragma unroll
            for (int kw = 0; kw < 4; ++kw) {
                int iw = ow * 2 - 1 + kw;
                if (iw < 0 || iw >= 256) continue;
                acc += x[(((size_t)b * 3 + ic) * 256 + ih) * 256 + iw] *
                       w[((oc * 3 + ic) * 4 + kh) * 4 + kw];
            }
        }
    }
    out[idx] = (_Float16)(acc > 0.f ? acc : 0.f);
}

// ---------------------------------------------------------------------------
// Implicit-GEMM conv via v_wmma_f32_16x16x32_f16.
// One wave -> 16 output pixels x 16 output channels.
// mode 0: relu -> f16 out16 ; mode 1: f32 store to outf ; mode 2: f32 += outf
// ---------------------------------------------------------------------------
template <int KH, int KW>
__global__ void conv_wmma_kernel(const _Float16* __restrict__ in16,
                                 const _Float16* __restrict__ w16,
                                 const float*    __restrict__ bias,
                                 _Float16* __restrict__ out16,
                                 float*    __restrict__ outf,
                                 int B, int Cin, int H, int W,
                                 int Cout, int OH, int OW,
                                 int stride, int pad, int mode)
{
    const int lane = threadIdx.x & 31;
    const int wave = blockIdx.x * (blockDim.x >> 5) + (threadIdx.x >> 5);
    const int Kdim   = Cin * KH * KW;
    const int ptiles = (OH * OW) >> 4;
    const int ctiles = Cout >> 4;
    const int tilesPerB = ptiles * ctiles;
    if (wave >= B * tilesPerB) return;

    int b  = wave / tilesPerB;
    int t  = wave - b * tilesPerB;
    int ct = t / ptiles;
    int pt = t - ct * ptiles;

    const int hi = lane >> 4;
    const int mn = lane & 15;
    const int p  = pt * 16 + mn;
    const int oh = p / OW;
    const int ow = p - oh * OW;
    const int oc = ct * 16 + mn;

    const _Float16* wrow = w16 + (size_t)oc * Kdim + hi * 16;

    v8f acc = {};
    for (int kbase = 0; kbase < Kdim; kbase += 32) {
        v16h bm = *(const v16h*)(wrow + kbase);
        if (kbase + 32 < Kdim)
            __builtin_prefetch(wrow + kbase + 32, 0, 1);

        v16h am;
#pragma unroll
        for (int e = 0; e < 16; ++e) {
            int k  = kbase + (hi << 4) + e;
            int ic = k / (KH * KW);
            int r  = k - ic * (KH * KW);
            int kh = r / KW;
            int kw = r - kh * KW;
            int ih = oh * stride - pad + kh;
            int iw = ow * stride - pad + kw;
            _Float16 v = (_Float16)0.f;
            if (ih >= 0 && ih < H && iw >= 0 && iw < W)
                v = in16[(((size_t)b * Cin + ic) * H + ih) * W + iw];
            am[e] = v;
        }
        acc = __builtin_amdgcn_wmma_f32_16x16x32_f16(
            false, am, false, bm, (short)0, acc, false, false);
    }

    const float bv = bias[oc];
#pragma unroll
    for (int i = 0; i < 8; ++i) {
        int m  = i + hi * 8;
        int pp = pt * 16 + m;
        size_t o = ((size_t)b * Cout + oc) * (size_t)(OH * OW) + pp;
        float v = acc[i] + bv;
        if (mode == 0)      out16[o] = (_Float16)(v > 0.f ? v : 0.f);
        else if (mode == 1) outf[o]  = v;
        else                outf[o] += v;
    }
}

// ---------------------------------------------------------------------------
// Elementwise helpers
// ---------------------------------------------------------------------------
__global__ void f32_to_f16_kernel(const float* __restrict__ s,
                                  _Float16* __restrict__ d, int n)
{
    for (int i = blockIdx.x * blockDim.x + threadIdx.x; i < n;
         i += gridDim.x * blockDim.x)
        d[i] = (_Float16)s[i];
}

__global__ void relu_f16_kernel(const float* __restrict__ s,
                                _Float16* __restrict__ d, int n)
{
    for (int i = blockIdx.x * blockDim.x + threadIdx.x; i < n;
         i += gridDim.x * blockDim.x) {
        float v = s[i];
        d[i] = (_Float16)(v > 0.f ? v : 0.f);
    }
}

__global__ void relu_f32_kernel(const float* __restrict__ s,
                                float* __restrict__ d, int n)
{
    for (int i = blockIdx.x * blockDim.x + threadIdx.x; i < n;
         i += gridDim.x * blockDim.x) {
        float v = s[i];
        d[i] = v > 0.f ? v : 0.f;
    }
}

__global__ void cbnorm_kernel(const float* __restrict__ cb,
                              float* __restrict__ cbn2)
{
    int k = blockIdx.x * blockDim.x + threadIdx.x;
    if (k >= 512) return;
    const float* row = cb + (size_t)k * 128;
    float s = 0.f;
#pragma unroll 8
    for (int d = 0; d < 128; ++d) s += row[d] * row[d];
    cbn2[k] = s;
}

// ---------------------------------------------------------------------------
// VQ argmin: one wave = 16 flat rows vs all 512 codewords (128 WMMAs).
// A rows staged via GLOBAL_LOAD_ASYNC_TO_LDS (ASYNCcnt), padded LDS stride
// 132 f32/row so the 16 same-k readers land on distinct banks.
// ---------------------------------------------------------------------------
#define VQ_WAVES 4
#define VQ_ROWSTRIDE 132                 // floats; 528B, 16B-aligned
#define VQ_SLAB (16 * VQ_ROWSTRIDE)      // 2112 floats per wave

__global__ void vq_argmin_kernel(const float* __restrict__ flat,
                                 const _Float16* __restrict__ cb16,
                                 const float* __restrict__ cbn2,
                                 int* __restrict__ idx_out, int rows)
{
    __shared__ float lds_a[VQ_WAVES * VQ_SLAB];
    const int lane  = threadIdx.x & 31;
    const int wslot = threadIdx.x >> 5;
    const int wave  = blockIdx.x * (blockDim.x >> 5) + wslot;
    const int rbase = wave * 16;
    if (rbase >= rows) return;
    const int hi = lane >> 4;
    const int mn = lane & 15;

    float* slab = &lds_a[wslot * VQ_SLAB];

#if defined(__AMDGCN__)
    {
        // one 512B row per issue: 32 lanes x 16B, async copy to LDS
        unsigned lbase = (unsigned)(unsigned long long)slab + (unsigned)(lane * 16);
        unsigned long long gbase =
            (unsigned long long)(const void*)(flat + (size_t)rbase * 128) +
            (unsigned long long)(lane * 16);
#pragma unroll
        for (int r = 0; r < 16; ++r) {
            unsigned loff = lbase + (unsigned)(r * (VQ_ROWSTRIDE * 4));
            unsigned long long gaddr = gbase + (unsigned long long)(r * 512);
            asm volatile("global_load_async_to_lds_b128 %0, %1, off"
                         :: "v"(loff), "v"(gaddr) : "memory");
        }
        asm volatile("s_wait_asynccnt 0x0" ::: "memory");
    }
#else
    for (int r = 0; r < 16; ++r)
        for (int d = lane; d < 128; d += 32)
            slab[r * VQ_ROWSTRIDE + d] = flat[(size_t)(rbase + r) * 128 + d];
#endif

    // A operand from LDS, contiguous k = 32*s + 16*hi + e, same order as B
    v16h a[4];
    const float* ar = slab + mn * VQ_ROWSTRIDE;
#pragma unroll
    for (int s = 0; s < 4; ++s) {
#pragma unroll
        for (int e = 0; e < 16; ++e) {
            int k = s * 32 + (hi << 4) + e;
            a[s][e] = (_Float16)ar[k];
        }
    }

    float runv[8];
    int   runi[8];
#pragma unroll
    for (int i = 0; i < 8; ++i) { runv[i] = 3.4e38f; runi[i] = 0x7fffffff; }

    for (int nt = 0; nt < 32; ++nt) {
        const int n0 = nt * 16;
        const _Float16* crow = cb16 + (size_t)(n0 + mn) * 128 + hi * 16;
        v8f acc = {};
#pragma unroll
        for (int s = 0; s < 4; ++s) {
            v16h bm = *(const v16h*)(crow + s * 32);
            acc = __builtin_amdgcn_wmma_f32_16x16x32_f16(
                false, a[s], false, bm, (short)0, acc, false, false);
        }
        const int   myn = n0 + mn;
        const float c2  = cbn2[myn];
#pragma unroll
        for (int i = 0; i < 8; ++i) {
            float d2 = c2 - 2.f * acc[i];
            if (d2 < runv[i]) { runv[i] = d2; runi[i] = myn; }
        }
    }

#pragma unroll
    for (int i = 0; i < 8; ++i) {
        float v  = runv[i];
        int   id = runi[i];
#pragma unroll
        for (int m = 1; m < 16; m <<= 1) {
            float ov = __shfl_xor(v, m, 32);
            int   oi = __shfl_xor(id, m, 32);
            if (ov < v || (ov == v && oi < id)) { v = ov; id = oi; }
        }
        runv[i] = v; runi[i] = id;
    }
    if (mn == 0) {
#pragma unroll
        for (int i = 0; i < 8; ++i)
            idx_out[rbase + i + hi * 8] = runi[i];
    }
}

// ---------------------------------------------------------------------------
// Output gather via TDM gather-mode: one wave gathers 16 codebook rows by
// 16-bit index (tensor_load_to_lds), then streams them out with
// global_store_async_from_lds. Fallback: plain float4 gather.
// ---------------------------------------------------------------------------
__global__ void gather_q_tdm_kernel(const int* __restrict__ idx,
                                    const float* __restrict__ cb,
                                    float* __restrict__ qout, int rows)
{
    __shared__ float lds_g[VQ_WAVES * 2048];      // 8KB per wave
    const int lane  = threadIdx.x & 31;
    const int wslot = threadIdx.x >> 5;
    const int wave  = blockIdx.x * (blockDim.x >> 5) + wslot;
    const int rbase = wave * 16;
    if (rbase >= rows) return;

#if defined(__gfx1250__) && __has_builtin(__builtin_amdgcn_tensor_load_to_lds)
    float* slab = &lds_g[wslot * 2048];
    unsigned lds_addr = (unsigned)(unsigned long long)slab;
    unsigned long long gaddr = (unsigned long long)(const void*)cb;

    // D# group 0: count=1, gather_mode=1 (16-bit idx), lds_addr, global_addr, type=2
    u32x4 g0;
    g0[0] = 0x80000001u;
    g0[1] = lds_addr;
    g0[2] = (unsigned)(gaddr & 0xffffffffu);
    g0[3] = (unsigned)((gaddr >> 32) & 0x1ffffffu) | (2u << 30);

    // D# group 1: data_size=4B, dim0=128, dim1=512, tile0=128, tile1=16 idx,
    // stride0=128
    i32x8 g1;
    g1[0] = 0x00020000;          // data_size=2 (4 bytes)
    g1[1] = (int)(128u << 16);   // tensor_dim0[15:0] at bits 31:16
    g1[2] = (int)(512u << 16);   // tensor_dim0 hi=0 | tensor_dim1[15:0]
    g1[3] = (int)(128u << 16);   // tensor_dim1 hi=0 | tile_dim0=128
    g1[4] = 16;                  // tile_dim1 = #valid indices
    g1[5] = 128;                 // tensor_dim0_stride lo
    g1[6] = 0;
    g1[7] = 0;

    // D# groups 2/3: 16 packed 16-bit row indices
    i32x4 g2, g3;
#pragma unroll
    for (int j = 0; j < 4; ++j)
        g2[j] = (idx[rbase + 2 * j] & 0xffff) |
                ((idx[rbase + 2 * j + 1] & 0xffff) << 16);
#pragma unroll
    for (int j = 0; j < 4; ++j)
        g3[j] = (idx[rbase + 8 + 2 * j] & 0xffff) |
                ((idx[rbase + 8 + 2 * j + 1] & 0xffff) << 16);

#if __clang_major__ >= 23
    i32x8 gz = {};
    __builtin_amdgcn_tensor_load_to_lds(g0, g1, g2, g3, gz, 0);
#else
    __builtin_amdgcn_tensor_load_to_lds(g0, g1, g2, g3, 0);
#endif
    __builtin_amdgcn_s_wait_tensorcnt((short)0);

    // stream the gathered 8KB to d_out: 16 issues x 32 lanes x 16B
    {
        unsigned lbase = lds_addr + (unsigned)(lane * 16);
        unsigned long long obase =
            (unsigned long long)(void*)(qout + (size_t)rbase * 128) +
            (unsigned long long)(lane * 16);
#pragma unroll
        for (int r = 0; r < 16; ++r) {
            unsigned loff = lbase + (unsigned)(r * 512);
            unsigned long long oaddr = obase + (unsigned long long)(r * 512);
            asm volatile("global_store_async_from_lds_b128 %0, %1, off"
                         :: "v"(oaddr), "v"(loff) : "memory");
        }
        asm volatile("s_wait_asynccnt 0x0" ::: "memory");
    }
#else
    (void)lds_g;
    for (int r = 0; r < 16; ++r) {
        int row = rbase + r;
        const float4* src = (const float4*)(cb + (size_t)idx[row] * 128);
        float4* dst = (float4*)(qout + (size_t)row * 128);
        dst[lane] = src[lane];
    }
#endif
}

__global__ void zero1_kernel(float* p)
{
    if (blockIdx.x == 0 && threadIdx.x == 0) *p = 0.f;
}

__global__ void commit_kernel(const float* __restrict__ q,
                              const float* __restrict__ f,
                              float* __restrict__ outs, int n, float inv_n)
{
    float s = 0.f;
    for (int i = blockIdx.x * blockDim.x + threadIdx.x; i < n;
         i += gridDim.x * blockDim.x) {
        float d = q[i] - f[i];
        s += d * d;
    }
#pragma unroll
    for (int m = 16; m > 0; m >>= 1) s += __shfl_xor(s, m, 32);
    if ((threadIdx.x & 31) == 0) atomicAdd(outs, s * inv_n);
}

// ---------------------------------------------------------------------------
extern "C" void kernel_launch(void* const* d_in, const int* in_sizes, int n_in,
                              void* d_out, int out_size, void* d_ws, size_t ws_size,
                              hipStream_t stream)
{
    (void)in_sizes; (void)n_in; (void)out_size; (void)ws_size;
    const float* x     = (const float*)d_in[0];
    const float* c0_w  = (const float*)d_in[1];
    const float* c0_b  = (const float*)d_in[2];
    const float* c1_w  = (const float*)d_in[3];
    const float* c1_b  = (const float*)d_in[4];
    const float* c2_w  = (const float*)d_in[5];
    const float* c2_b  = (const float*)d_in[6];
    const float* c3_w  = (const float*)d_in[7];
    const float* c3_b  = (const float*)d_in[8];
    const float* r0a_w = (const float*)d_in[9];
    const float* r0a_b = (const float*)d_in[10];
    const float* r0b_w = (const float*)d_in[11];
    const float* r0b_b = (const float*)d_in[12];
    const float* r1a_w = (const float*)d_in[13];
    const float* r1a_b = (const float*)d_in[14];
    const float* r1b_w = (const float*)d_in[15];
    const float* r1b_b = (const float*)d_in[16];
    const float* cb    = (const float*)d_in[17];
    float* out = (float*)d_out;

    char* ws = (char*)d_ws;
    size_t off = 0;
    auto alloc = [&](size_t bytes) -> void* {
        void* p = ws + off;
        off += (bytes + 255) & ~(size_t)255;
        return p;
    };
    _Float16* h0    = (_Float16*)alloc((size_t)33554432 * 2); // 64x32x128x128
    _Float16* h1    = (_Float16*)alloc((size_t)16777216 * 2); // 64x64x64x64
    _Float16* h2    = (_Float16*)alloc((size_t)8388608  * 2); // 64x128x32x32
    float*    x3    = (float*)   alloc((size_t)8388608  * 4); // residual f32
    _Float16* t16   = (_Float16*)alloc((size_t)8388608  * 2); // relu(x3) f16
    _Float16* w16c1 = (_Float16*)alloc((size_t)32768  * 2);
    _Float16* w16c2 = (_Float16*)alloc((size_t)131072 * 2);
    _Float16* w16c3 = (_Float16*)alloc((size_t)147456 * 2);
    _Float16* w16r0a= (_Float16*)alloc((size_t)36864  * 2);
    _Float16* w16r0b= (_Float16*)alloc((size_t)4096   * 2);
    _Float16* w16r1a= (_Float16*)alloc((size_t)36864  * 2);
    _Float16* w16r1b= (_Float16*)alloc((size_t)4096   * 2);
    _Float16* cb16  = (_Float16*)alloc((size_t)65536  * 2);
    float*    cbn2  = (float*)   alloc((size_t)512    * 4);
    int*      idx   = (int*)     alloc((size_t)65536  * 4);
    float*    flat  = (float*)h0;   // alias: h0 dead before flat is written

    const int T = 256;
    auto gsz = [](int n, int b) { return (n + b - 1) / b; };

    f32_to_f16_kernel<<<gsz(32768,  T), T, 0, stream>>>(c1_w,  w16c1,  32768);
    f32_to_f16_kernel<<<gsz(131072, T), T, 0, stream>>>(c2_w,  w16c2,  131072);
    f32_to_f16_kernel<<<gsz(147456, T), T, 0, stream>>>(c3_w,  w16c3,  147456);
    f32_to_f16_kernel<<<gsz(36864,  T), T, 0, stream>>>(r0a_w, w16r0a, 36864);
    f32_to_f16_kernel<<<gsz(4096,   T), T, 0, stream>>>(r0b_w, w16r0b, 4096);
    f32_to_f16_kernel<<<gsz(36864,  T), T, 0, stream>>>(r1a_w, w16r1a, 36864);
    f32_to_f16_kernel<<<gsz(4096,   T), T, 0, stream>>>(r1b_w, w16r1b, 4096);
    f32_to_f16_kernel<<<gsz(65536,  T), T, 0, stream>>>(cb,    cb16,   65536);
    cbnorm_kernel<<<2, 256, 0, stream>>>(cb, cbn2);

    conv0_kernel<<<gsz(33554432, T), T, 0, stream>>>(x, c0_w, c0_b, h0);

    conv_wmma_kernel<4,4><<<8192, T, 0, stream>>>(h0, w16c1, c1_b, h1, (float*)0,
        64, 32, 128, 128, 64, 64, 64, 2, 1, 0);
    conv_wmma_kernel<4,4><<<4096, T, 0, stream>>>(h1, w16c2, c2_b, h2, (float*)0,
        64, 64, 64, 64, 128, 32, 32, 2, 1, 0);
    conv_wmma_kernel<3,3><<<4096, T, 0, stream>>>(h2, w16c3, c3_b, (_Float16*)0, x3,
        64, 128, 32, 32, 128, 32, 32, 1, 1, 1);

    relu_f16_kernel<<<4096, T, 0, stream>>>(x3, t16, 8388608);
    conv_wmma_kernel<3,3><<<1024, T, 0, stream>>>(t16, w16r0a, r0a_b, h2, (float*)0,
        64, 128, 32, 32, 32, 32, 32, 1, 1, 0);
    conv_wmma_kernel<1,1><<<4096, T, 0, stream>>>(h2, w16r0b, r0b_b, (_Float16*)0, x3,
        64, 32, 32, 32, 128, 32, 32, 1, 0, 2);
    relu_f16_kernel<<<4096, T, 0, stream>>>(x3, t16, 8388608);
    conv_wmma_kernel<3,3><<<1024, T, 0, stream>>>(t16, w16r1a, r1a_b, h2, (float*)0,
        64, 128, 32, 32, 32, 32, 32, 1, 1, 0);
    conv_wmma_kernel<1,1><<<4096, T, 0, stream>>>(h2, w16r1b, r1b_b, (_Float16*)0, x3,
        64, 32, 32, 32, 128, 32, 32, 1, 0, 2);

    relu_f32_kernel<<<4096, T, 0, stream>>>(x3, flat, 8388608);

    // VQ: 4096 waves, 4 waves/block (33KB LDS staging per block)
    vq_argmin_kernel<<<1024, 128, 0, stream>>>(flat, cb16, cbn2, idx, 65536);
    // output gather via TDM gather-mode + async LDS->global store
    gather_q_tdm_kernel<<<1024, 128, 0, stream>>>(idx, cb, out, 65536);
    zero1_kernel<<<1, 1, 0, stream>>>(out + 8388608);
    commit_kernel<<<4096, T, 0, stream>>>(out, flat, out + 8388608,
                                          8388608, 1.0f / 8388608.0f);
}
